// PCPolicy_6047313952963
// MI455X (gfx1250) — matrix operand order repaired and checked by reading, not compiled, not run on hardware
//
#include <hip/hip_runtime.h>
#include <hip/hip_bf16.h>

typedef __attribute__((ext_vector_type(16))) __bf16 v16bf;
typedef __attribute__((ext_vector_type(8)))  float  v8f;
typedef __attribute__((ext_vector_type(4)))  float  v4f;

#define B_  4
#define N_  512
#define NF_ 16
#define EF_ 16
#define E_  128
#define L_  3

__device__ __forceinline__ __bf16 to_bf16(float x) { return (__bf16)x; }

__device__ __forceinline__ float fast_tanh(float x) {
#if __has_builtin(__builtin_amdgcn_tanhf)
  return __builtin_amdgcn_tanhf(x);          // v_tanh_f32 (gfx1250 TRANS op)
#else
  // tanh(x) = 1 - 2/(exp(2x)+1); exp(2x) = exp2(x * 2*log2(e))
  float e = __builtin_amdgcn_exp2f(x * 2.8853900817779268f);
  return 1.0f - 2.0f * __builtin_amdgcn_rcpf(e + 1.0f);
#endif
}

__device__ __forceinline__ float fast_log(float x) {
  return __builtin_amdgcn_logf(x) * 0.6931471805599453f;
}

// ---------------------------------------------------------------------------
// Kernel 1: per-(b,i) row. Embed -> 3 message-passing layers -> h_out.
// Block = 128 threads = 4 waves; wave w owns e-columns [16w,16w+16) and
// [16w+64, 16w+80) -> one A build feeds TWO back-to-back WMMAs.
// Pre-tanh message tile via v_wmma_f32_16x16x32_bf16 with
//   A = [ef(16) | W_ij | 1 | 0...]  (16 j-rows x 32 K)
//   B = [We ; w_l ; hi_i ; 0...]    (32 K x 16 e-cols)
// ---------------------------------------------------------------------------
__global__ __launch_bounds__(128)
void mp_fused_kernel(const float* __restrict__ node_features,   // (B,N,16)
                     const float* __restrict__ edge_features,   // (B,N,N,16)
                     const float* __restrict__ Wmat,            // (B,N,N)
                     const float* __restrict__ build_limit,     // (B,3)
                     const float* __restrict__ open_group_size, // (B,3)
                     const int*   __restrict__ assigned,        // (B,N)
                     const int*   __restrict__ open_group,      // (B,N)
                     const int*   __restrict__ action_mask,     // (B,N)
                     const int*   __restrict__ fallback_mask,   // (B,N)
                     const float* __restrict__ node_embed_w,    // (23,128)
                     const float* __restrict__ node_embed_b,    // (128)
                     const float* __restrict__ mp_node_w,       // (3,128,128)
                     const float* __restrict__ mp_edge_w,       // (3,16,128)
                     const float* __restrict__ mp_weight_w,     // (3,128)
                     const float* __restrict__ mp_out_w,        // (3,128,128)
                     const float* __restrict__ mp_out_b,        // (3,128)
                     float*       __restrict__ h_out)           // (B,N,128)
{
  __shared__ float sH[E_];
  __shared__ float sHi[E_];
  __shared__ float sAgg[E_];

  const int bi   = blockIdx.x;
  const int b    = bi >> 9;
  const int i    = bi & (N_ - 1);
  const int tid  = threadIdx.x;
  const int lane = tid & 31;
  const int wave = tid >> 5;
  const int e0   = wave * 16;        // first e-tile
  const int e1   = e0 + 64;          // second e-tile

  // ---- embed: h0 = [nf | dyn] @ node_embed_w + b ----
  {
    float acc = node_embed_b[tid];
    const float* nf = node_features + (size_t)(b * N_ + i) * NF_;
    #pragma unroll
    for (int f = 0; f < NF_; ++f)
      acc += nf[f] * node_embed_w[f * E_ + tid];
    float dyn[7];
    dyn[0] = (float)(assigned[b * N_ + i] != 0);
    dyn[1] = (float)(open_group[b * N_ + i] != 0);
    dyn[2] = (float)(action_mask[b * N_ + i] != 0);
    dyn[3] = (float)(fallback_mask[b * N_ + i] != 0);
    #pragma unroll
    for (int k = 0; k < 3; ++k)
      dyn[4 + k] = open_group_size[b * 3 + k] / fmaxf(build_limit[b * 3 + k], 1e-6f);
    #pragma unroll
    for (int k = 0; k < 7; ++k)
      acc += dyn[k] * node_embed_w[(NF_ + k) * E_ + tid];
    sH[tid] = acc;
  }
  __syncthreads();

  const float* efb = edge_features + (size_t)(b * N_ + i) * N_ * EF_;
  const float* wb  = Wmat + (size_t)(b * N_ + i) * N_;

  // Static parts of the A operand (same for all layers / j-tiles):
  // lanes 0-15:  elems 8..15 = K16..23 = {W_ij, 1, 0,0,0,0,0,0}
  // lanes 16-31: elems 8..15 = K24..31 = all zero
  v16bf am;
  #pragma unroll
  for (int k = 0; k < 16; ++k) am[k] = to_bf16(0.0f);
  if (lane < 16) am[9] = to_bf16(1.0f);

  for (int l = 0; l < L_; ++l) {
    // hi = h @ mp_node_w[l]
    {
      const float* wn = mp_node_w + (size_t)l * E_ * E_;
      float acc = 0.f;
      for (int f = 0; f < E_; ++f)
        acc += sH[f] * wn[f * E_ + tid];
      sHi[tid] = acc;
    }
    __syncthreads();

    // ---- build the two B operands (32 K x 16 N each) ----
    // lanes 0-15:  N = col, elems 0..7 = We[l][0..8)[col], elems 8..9 = {w_l[col], hi[col]}, rest 0
    // lanes 16-31: N = col, elems 0..7 = We[l][8..16)[col], elems 8..15 = 0
    v16bf bm0, bm1;
    {
      const int c0 = e0 + (lane & 15);
      const int c1 = e1 + (lane & 15);
      const float* we = mp_edge_w + (size_t)l * EF_ * E_;
      const int kb = (lane >> 4) * 8;
      #pragma unroll
      for (int k = 0; k < 8; ++k) bm0[k] = to_bf16(we[(kb + k) * E_ + c0]);
      #pragma unroll
      for (int k = 0; k < 8; ++k) bm1[k] = to_bf16(we[(kb + k) * E_ + c1]);
      #pragma unroll
      for (int k = 8; k < 16; ++k) { bm0[k] = to_bf16(0.0f); bm1[k] = to_bf16(0.0f); }
      if (lane < 16) {
        bm0[8] = to_bf16(mp_weight_w[l * E_ + c0]);
        bm0[9] = to_bf16(sHi[c0]);
        bm1[8] = to_bf16(mp_weight_w[l * E_ + c1]);
        bm1[9] = to_bf16(sHi[c1]);
      }
    }

    float acc0 = 0.f, acc1 = 0.f;
    for (int j0 = 0; j0 < N_; j0 += 16) {
      // ---- dynamic part of A (16 M x 32 K) ----
      // lanes 0-15:  M = lane,    elems 0..7 = ef[0..8),  elem 8 = W_ij
      // lanes 16-31: M = lane-16, elems 0..7 = ef[8..16)
      const float* efrow = efb + (size_t)(j0 + (lane & 15)) * EF_ + ((lane >> 4) ? 8 : 0);
      v4f f0 = *(const v4f*)(efrow);
      v4f f1 = *(const v4f*)(efrow + 4);
      if (j0 + 16 < N_) __builtin_prefetch(efrow + 16 * EF_, 0, 3);

      #pragma unroll
      for (int k = 0; k < 4; ++k) am[k]     = to_bf16(f0[k]);
      #pragma unroll
      for (int k = 0; k < 4; ++k) am[4 + k] = to_bf16(f1[k]);
      if (lane < 16) am[8] = to_bf16(wb[j0 + lane]);

      v8f c = {};
      v8f d0 = __builtin_amdgcn_wmma_f32_16x16x32_bf16(
          false, am, false, bm0, (short)0, c, false, false);
      v8f d1 = __builtin_amdgcn_wmma_f32_16x16x32_bf16(
          false, am, false, bm1, (short)0, c, false, false);
      #pragma unroll
      for (int k = 0; k < 8; ++k) acc0 += fast_tanh(d0[k]);
      #pragma unroll
      for (int k = 0; k < 8; ++k) acc1 += fast_tanh(d1[k]);
    }

    // lanes L and L+16 cover M rows 0-7 / 8-15 of the same column -> pair-sum
    acc0 += __shfl_xor(acc0, 16, 32);
    acc1 += __shfl_xor(acc1, 16, 32);
    if (lane < 16) {
      sAgg[e0 + lane] = acc0 * (1.0f / (float)N_);
      sAgg[e1 + lane] = acc1 * (1.0f / (float)N_);
    }
    __syncthreads();

    // h = h + agg @ mp_out_w[l] + mp_out_b[l]
    float newh;
    {
      const float* wo = mp_out_w + (size_t)l * E_ * E_;
      float a2 = mp_out_b[l * E_ + tid];
      for (int f = 0; f < E_; ++f)
        a2 += sAgg[f] * wo[f * E_ + tid];
      newh = sH[tid] + a2;
    }
    __syncthreads();
    sH[tid] = newh;
    __syncthreads();
  }

  h_out[(size_t)(b * N_ + i) * E_ + tid] = sH[tid];
}

// ---------------------------------------------------------------------------
// Kernel 2: per-batch pooling + attention + softmax + argmax/logp/entropy.
// out layout: [4 action][4 logp][4 entropy][4*512 logits]
// ---------------------------------------------------------------------------
__global__ __launch_bounds__(512)
void head_kernel(const float* __restrict__ h,           // (B,N,128)
                 const int*   __restrict__ open_group,  // (B,N)
                 const int*   __restrict__ action_mask, // (B,N)
                 const float* __restrict__ context_w,   // (128,128)
                 const float* __restrict__ key_w,       // (128,128)
                 const float* __restrict__ sep_gate_w,  // (128,1)
                 const float* __restrict__ sep_gate_b,  // (1)
                 const float* __restrict__ logit_bias,  // (1)
                 float*       __restrict__ out)
{
  const int b   = blockIdx.x;
  const int tid = threadIdx.x;
  __shared__ float sCtx[E_];
  __shared__ float sT[E_];
  __shared__ float sLog[N_];
  __shared__ float sRed[N_];
  __shared__ int   sRedI[N_];
  __shared__ float sOgDen;
  __shared__ int   sHasOpen;
  __shared__ float sMax, sSum;
  __shared__ int   sArg;

  const float* hb = h + (size_t)b * N_ * E_;
  const int*   og = open_group + b * N_;

  if (tid == 0) {
    int s = 0, any = 0;
    for (int n = 0; n < N_; ++n) { int v = (og[n] != 0); s += v; any |= v; }
    sOgDen = fmaxf((float)s, 1.0f);
    sHasOpen = any;
  }
  __syncthreads();

  // context = has_open ? mean of open-group rows : h[b,0,:]
  if (tid < E_) {
    float gm = 0.f;
    for (int n = 0; n < N_; ++n)
      if (og[n]) gm += hb[(size_t)n * E_ + tid];
    gm /= sOgDen;
    sCtx[tid] = sHasOpen ? gm : hb[tid];
  }
  __syncthreads();

  // query = context @ context_w  (stashed in sRed)
  if (tid < E_) {
    float q = 0.f;
    for (int f = 0; f < E_; ++f) q += sCtx[f] * context_w[f * E_ + tid];
    sRed[tid] = q;
  }
  __syncthreads();
  // t[f] = sum_e key_w[f][e] * query[e]  =>  logits_n = h[n,:].t
  if (tid < E_) {
    float t = 0.f;
    for (int e = 0; e < E_; ++e) t += key_w[tid * E_ + e] * sRed[e];
    sT[tid] = t;
  }
  __syncthreads();

  // logits
  {
    const int n = tid;
    float lg = 0.f;
    for (int f = 0; f < E_; ++f) lg += hb[(size_t)n * E_ + f] * sT[f];
    lg *= 0.08838834764831845f; // 1/sqrt(128)
    if (n == 0) {
      float s = sep_gate_b[0];
      for (int f = 0; f < E_; ++f) s += sCtx[f] * sep_gate_w[f];
      lg += s;
    }
    lg += logit_bias[0];
    lg *= (1.0f / 1.2f);
    if (!action_mask[b * N_ + n]) lg = -1e9f;
    sLog[n] = lg;
  }
  __syncthreads();

  // argmax (first index on ties) + max
  sRed[tid] = sLog[tid]; sRedI[tid] = tid;
  __syncthreads();
  for (int s = 256; s > 0; s >>= 1) {
    if (tid < s) {
      float a = sRed[tid], bv = sRed[tid + s];
      int   ai = sRedI[tid], bi2 = sRedI[tid + s];
      if (bv > a || (bv == a && bi2 < ai)) { sRed[tid] = bv; sRedI[tid] = bi2; }
    }
    __syncthreads();
  }
  if (tid == 0) { sMax = sRed[0]; sArg = sRedI[0]; }
  __syncthreads();

  float ex = __builtin_amdgcn_exp2f((sLog[tid] - sMax) * 1.4426950408889634f);
  sRed[tid] = ex;
  __syncthreads();
  for (int s = 256; s > 0; s >>= 1) {
    if (tid < s) sRed[tid] += sRed[tid + s];
    __syncthreads();
  }
  if (tid == 0) sSum = sRed[0];
  __syncthreads();

  float p  = ex / sSum;
  float lp = fast_log(fmaxf(p, 1e-12f));
  sRed[tid] = -p * lp;
  __syncthreads();
  for (int s = 256; s > 0; s >>= 1) {
    if (tid < s) sRed[tid] += sRed[tid + s];
    __syncthreads();
  }

  if (tid == 0) {
    out[b] = (float)sArg;
    float pe = __builtin_amdgcn_exp2f((sLog[sArg] - sMax) * 1.4426950408889634f) / sSum;
    out[4 + b] = fast_log(fmaxf(pe, 1e-12f));
    out[8 + b] = sRed[0];
  }
  out[12 + b * N_ + tid] = sLog[tid];
}

extern "C" void kernel_launch(void* const* d_in, const int* in_sizes, int n_in,
                              void* d_out, int out_size, void* d_ws, size_t ws_size,
                              hipStream_t stream) {
  const float* node_features   = (const float*)d_in[0];
  const float* edge_features   = (const float*)d_in[1];
  const float* Wmat            = (const float*)d_in[2];
  const float* build_limit     = (const float*)d_in[3];
  const float* open_group_size = (const float*)d_in[4];
  const int*   assigned        = (const int*)d_in[5];
  const int*   open_group      = (const int*)d_in[6];
  const int*   action_mask     = (const int*)d_in[7];
  const int*   fallback_mask   = (const int*)d_in[8];
  const float* node_embed_w    = (const float*)d_in[9];
  const float* node_embed_b    = (const float*)d_in[10];
  const float* mp_node_w       = (const float*)d_in[11];
  const float* mp_edge_w       = (const float*)d_in[12];
  const float* mp_weight_w     = (const float*)d_in[13];
  const float* mp_out_w        = (const float*)d_in[14];
  const float* mp_out_b        = (const float*)d_in[15];
  const float* context_w       = (const float*)d_in[16];
  const float* key_w           = (const float*)d_in[17];
  const float* sep_gate_w      = (const float*)d_in[18];
  const float* sep_gate_b      = (const float*)d_in[19];
  const float* logit_bias      = (const float*)d_in[20];

  float* h_ws = (float*)d_ws; // (B,N,128) = 1 MiB
  float* out  = (float*)d_out;

  mp_fused_kernel<<<B_ * N_, 128, 0, stream>>>(
      node_features, edge_features, Wmat, build_limit, open_group_size,
      assigned, open_group, action_mask, fallback_mask,
      node_embed_w, node_embed_b, mp_node_w, mp_edge_w, mp_weight_w,
      mp_out_w, mp_out_b, h_ws);

  head_kernel<<<B_, 512, 0, stream>>>(
      h_ws, open_group, action_mask, context_w, key_w,
      sep_gate_w, sep_gate_b, logit_bias, out);
}
